// SelectiveSSM_41669772706451
// MI455X (gfx1250) — matrix operand order repaired and checked by reading, compiled, not run on hardware
//
#include <hip/hip_runtime.h>
#include <math.h>

// ---------------------------------------------------------------------------
// Types for CDNA5 WMMA (wave32): D(f32 16x16) = A(bf16 16x32) x B(bf16 32x16) + C
// ---------------------------------------------------------------------------
typedef __attribute__((ext_vector_type(16))) __bf16 bf16x16;
typedef __attribute__((ext_vector_type(4)))  __bf16 bf16x4;
typedef __attribute__((ext_vector_type(8)))  float  f32x8;
typedef __attribute__((ext_vector_type(4)))  float  f32x4;

#define BM 128
#define BN 64
#define BK 32
#define STRIDE 40   // LDS row stride in bf16 elements (32 data + 8 pad, keeps 16B align)

union FragU { uint4 q[2]; bf16x16 v; };

__device__ __forceinline__ bf16x4 cvt4(float4 f) {
  f32x4 v = {f.x, f.y, f.z, f.w};
  return __builtin_convertvector(v, bf16x4);   // v_cvt_pk_bf16_f32 (RNE)
}

// A-matrix 16x32 bf16 fragment (ISA 7.12.2): lane group g (lane>>4):
//   v0..v3 hold K = g*8 .. g*8+7, v4..v7 hold K = 16+g*8 .. 23+g*8, row M = lane&15.
__device__ __forceinline__ bf16x16 load_fragA(const __bf16* lds, int row0, int lane) {
  int r = row0 + (lane & 15);
  int g = (lane >> 4) << 3;                 // 0 or 8
  const __bf16* p = lds + r * STRIDE + g;
  FragU f;
  f.q[0] = *reinterpret_cast<const uint4*>(p);        // K = g .. g+7
  f.q[1] = *reinterpret_cast<const uint4*>(p + 16);   // K = 16+g .. 23+g
  return f.v;
}

// B-matrix 32x16 bf16 fragment: lane holds column n = lane&15; lane group g holds
// K = g*8 .. g*8+7 in v0..v3 and K = 16+g*8 .. 23+g*8 in v4..v7. LDS tile stored transposed: sB[n][k].
__device__ __forceinline__ bf16x16 load_fragB(const __bf16* lds, int col0, int lane) {
  int n = col0 + (lane & 15);
  int g = (lane >> 4) << 3;                 // 0 or 8
  const __bf16* p = lds + n * STRIDE + g;
  FragU f;
  f.q[0] = *reinterpret_cast<const uint4*>(p);        // K = g .. g+7
  f.q[1] = *reinterpret_cast<const uint4*>(p + 16);   // K = 16+g .. 23+g
  return f.v;
}

__device__ __forceinline__ void wmma_block(const __bf16* sA, const __bf16* sB,
                                           int wm, int wn, int lane, f32x8 acc[2][2]) {
  bf16x16 a0 = load_fragA(sA, wm * 32 + 0,  lane);
  bf16x16 a1 = load_fragA(sA, wm * 32 + 16, lane);
  bf16x16 b0 = load_fragB(sB, wn * 32 + 0,  lane);
  bf16x16 b1 = load_fragB(sB, wn * 32 + 16, lane);
  acc[0][0] = __builtin_amdgcn_wmma_f32_16x16x32_bf16(false, a0, false, b0, (short)0, acc[0][0], false, false);
  acc[0][1] = __builtin_amdgcn_wmma_f32_16x16x32_bf16(false, a0, false, b1, (short)0, acc[0][1], false, false);
  acc[1][0] = __builtin_amdgcn_wmma_f32_16x16x32_bf16(false, a1, false, b0, (short)0, acc[1][0], false, false);
  acc[1][1] = __builtin_amdgcn_wmma_f32_16x16x32_bf16(false, a1, false, b1, (short)0, acc[1][1], false, false);
}

// ---------------------------------------------------------------------------
// Generic GEMM: C[M,N] = A[M,K] @ W[N,K]^T (+bias), fp32 -> bf16 staging.
// Requires M % BM == 0 (true for all launches here).
// NGUARD=false requires N % BN == 0 (straight-line staging, no branches).
// MODE 0: plain (optional bias). MODE 2: dt epilogue sigmoid -> clip[1e-4, 1].
// ---------------------------------------------------------------------------
template <int MODE, bool NGUARD>
__global__ void __launch_bounds__(256)
gemm_kernel(const float* __restrict__ A, const float* __restrict__ W,
            float* __restrict__ C, const float* __restrict__ bias,
            int M, int N, int K, int lda, int ldout) {
  __shared__ __bf16 sA[BM * STRIDE];
  __shared__ __bf16 sB[BN * STRIDE];
  const int tid = threadIdx.x, lane = tid & 31, wave = tid >> 5;
  const int wm = wave & 3, wn = wave >> 2;
  const int m0 = blockIdx.x * BM, n0 = blockIdx.y * BN;

  const f32x8 zero = {0.f, 0.f, 0.f, 0.f, 0.f, 0.f, 0.f, 0.f};
  f32x8 acc[2][2] = {{zero, zero}, {zero, zero}};

  for (int k0 = 0; k0 < K; k0 += BK) {
    __syncthreads();
    // ---- stage A: 128x32, float4 loads, packed cvt to bf16 ----
#pragma unroll
    for (int i = 0; i < 4; ++i) {
      int c = tid + (i << 8);                // 0..1023
      int row = c >> 3, col4 = (c & 7) << 2;
      const float* src = A + (size_t)(m0 + row) * lda + k0 + col4;
      float4 af = *reinterpret_cast<const float4*>(src);
      if (k0 + BK < K) __builtin_prefetch(src + BK, 0, 1);   // global_prefetch_b8
      *reinterpret_cast<bf16x4*>(&sA[row * STRIDE + col4]) = cvt4(af);
    }
    // ---- stage W tile transposed: sB[n][k], 64x32 ----
#pragma unroll
    for (int i = 0; i < 2; ++i) {
      int c = tid + (i << 8);                // 0..511
      int n = c >> 3, col4 = (c & 7) << 2;
      int gn = n0 + n;
      float4 wf = {0.f, 0.f, 0.f, 0.f};
      if (!NGUARD || gn < N) {
        const float* src = W + (size_t)gn * K + k0 + col4;
        wf = *reinterpret_cast<const float4*>(src);
        if (k0 + BK < K) __builtin_prefetch(src + BK, 0, 1);
      }
      *reinterpret_cast<bf16x4*>(&sB[n * STRIDE + col4]) = cvt4(wf);
    }
    __syncthreads();
    wmma_block(sA, sB, wm, wn, lane, acc);
  }

  // ---- epilogue / store (C/D layout: vgpr r -> row r + 8*(lane>=16), col = lane&15) ----
#pragma unroll
  for (int ti = 0; ti < 2; ++ti)
#pragma unroll
    for (int tj = 0; tj < 2; ++tj)
#pragma unroll
      for (int r = 0; r < 8; ++r) {
        int grow = m0 + wm * 32 + ti * 16 + ((lane >> 4) << 3) + r;
        int gcol = n0 + wn * 32 + tj * 16 + (lane & 15);
        if (!NGUARD || gcol < N) {
          float v = acc[ti][tj][r];
          if (bias) v += bias[gcol];
          if constexpr (MODE == 2) {
            v = 1.f / (1.f + __expf(-v));                 // sigmoid
            v = fminf(fmaxf(v, 1e-4f), 1.0f);             // clip
          }
          C[(size_t)grow * ldout + gcol] = v;
        }
      }
}

// ---------------------------------------------------------------------------
// Causal full-channel conv1d as 4 accumulated GEMM passes over shifted rows.
// u[b,l,o] = conv_b[o] + sum_{t=0..3} sum_i xc[b, l-3+t, i] * conv_w[o, i, t]
// xc = xz columns [0, 2048), lda(xz) = 4096. conv_w layout [O=2048, I=2048, T=4].
// ---------------------------------------------------------------------------
__global__ void __launch_bounds__(256)
conv_gemm_kernel(const float* __restrict__ xz, const float* __restrict__ convw,
                 const float* __restrict__ convb, float* __restrict__ U) {
  __shared__ __bf16 sA[BM * STRIDE];
  __shared__ __bf16 sB[BN * STRIDE];
  const int tid = threadIdx.x, lane = tid & 31, wave = tid >> 5;
  const int wm = wave & 3, wn = wave >> 2;
  const int m0 = blockIdx.x * BM, n0 = blockIdx.y * BN;

  const f32x8 zero = {0.f, 0.f, 0.f, 0.f, 0.f, 0.f, 0.f, 0.f};
  f32x8 acc[2][2] = {{zero, zero}, {zero, zero}};

  for (int kc = 0; kc < 4; ++kc) {
    for (int k0 = 0; k0 < 2048; k0 += BK) {
      __syncthreads();
      // A: rows shifted by (kc-3) within each batch, zero-padded on the left
#pragma unroll
      for (int i = 0; i < 4; ++i) {
        int c = tid + (i << 8);
        int row = c >> 3, col4 = (c & 7) << 2;
        int gm = m0 + row;
        int b = gm >> 11, l = gm & 2047;
        int ls = l - 3 + kc;
        float4 af = {0.f, 0.f, 0.f, 0.f};
        if (ls >= 0)
          af = *reinterpret_cast<const float4*>(xz + ((size_t)(b * 2048 + ls)) * 4096 + k0 + col4);
        *reinterpret_cast<bf16x4*>(&sA[row * STRIDE + col4]) = cvt4(af);
      }
      // W_k[n,c] = conv_w[(n*2048 + c)*4 + kc]  (stride-4 gather, L2 resident)
#pragma unroll
      for (int i = 0; i < 8; ++i) {
        int e = (tid << 3) + i;                 // 0..2047
        int n = e >> 5, cc = e & 31;
        int gn = n0 + n, gc = k0 + cc;
        sB[n * STRIDE + cc] = (__bf16)convw[((size_t)gn * 2048 + gc) * 4 + kc];
      }
      __syncthreads();
      wmma_block(sA, sB, wm, wn, lane, acc);
    }
  }

#pragma unroll
  for (int ti = 0; ti < 2; ++ti)
#pragma unroll
    for (int tj = 0; tj < 2; ++tj)
#pragma unroll
      for (int r = 0; r < 8; ++r) {
        int grow = m0 + wm * 32 + ti * 16 + ((lane >> 4) << 3) + r;
        int gcol = n0 + wn * 32 + tj * 16 + (lane & 15);
        U[(size_t)grow * 2048 + gcol] = acc[ti][tj][r] + convb[gcol];
      }
}

// ---------------------------------------------------------------------------
// Selective scan: thread = one (batch, channel). 16-wide state in registers.
// B/C [l, 16] broadcast through LDS in 64-step chunks. Fuses SiLU gate and
// writes y over the dead xc half of the xz buffer (cols [0,2048), ld 4096).
// ---------------------------------------------------------------------------
__global__ void __launch_bounds__(256)
scan_kernel(const float* __restrict__ U, const float* __restrict__ DT,
            const float* __restrict__ Bm, const float* __restrict__ Cm,
            const float* __restrict__ A_log, const float* __restrict__ Dv,
            float* __restrict__ xz) {
  __shared__ float sB[64 * 16];
  __shared__ float sC[64 * 16];
  const int tid = threadIdx.x;
  const int b = blockIdx.x >> 3;                 // 2 batches x 8 channel blocks
  const int d = ((blockIdx.x & 7) << 8) + tid;   // channel 0..2047

  float Arow[16], state[16];
#pragma unroll
  for (int n = 0; n < 16; ++n) {
    Arow[n] = -__expf(A_log[d * 16 + n]);        // A = -exp(A_log)
    state[n] = 0.f;
  }
  const float Dd = Dv[d];

  for (int l0 = 0; l0 < 2048; l0 += 64) {
    __syncthreads();
    const size_t base = ((size_t)b * 2048 + l0) * 16;
#pragma unroll
    for (int i = 0; i < 4; ++i) {
      int idx = tid + (i << 8);                  // 1024 = 64*16
      sB[idx] = Bm[base + idx];
      sC[idx] = Cm[base + idx];
    }
    __syncthreads();
    for (int lc = 0; lc < 64; ++lc) {
      const size_t m = (size_t)b * 2048 + l0 + lc;
      float uv  = U[m * 2048 + d];
      float dtv = DT[m * 2048 + d];
      float zv  = xz[m * 4096 + 2048 + d];       // z half
      float accy = 0.f;
#pragma unroll
      for (int n = 0; n < 16; ++n) {
        float t = dtv * Arow[n];
        t = fminf(fmaxf(t, -5.f), 5.f);
        float da = __expf(t);
        state[n] = da * state[n] + dtv * sB[lc * 16 + n] * uv + 1e-6f;
        accy += state[n] * sC[lc * 16 + n];
      }
      float yv = accy + Dd * uv;
      float sig = 1.f / (1.f + __expf(-zv));
      xz[m * 4096 + d] = yv * zv * sig;          // y * silu(z) over dead xc slot
    }
  }
}

// ---------------------------------------------------------------------------
// Orchestration
// ---------------------------------------------------------------------------
extern "C" void kernel_launch(void* const* d_in, const int* in_sizes, int n_in,
                              void* d_out, int out_size, void* d_ws, size_t ws_size,
                              hipStream_t stream) {
  (void)in_sizes; (void)n_in; (void)out_size; (void)ws_size;
  const float* x        = (const float*)d_in[0];   // [2,2048,1024]
  const float* in_projw = (const float*)d_in[1];   // [4096,1024]
  const float* conv_w   = (const float*)d_in[2];   // [2048,2048,4]
  const float* conv_b   = (const float*)d_in[3];   // [2048]
  const float* dt_w     = (const float*)d_in[4];   // [2048,2048]
  const float* dt_b     = (const float*)d_in[5];   // [2048]
  const float* A_log    = (const float*)d_in[6];   // [2048,16]
  const float* Dv       = (const float*)d_in[7];   // [2048]
  const float* B_w      = (const float*)d_in[8];   // [16,2048]
  const float* C_w      = (const float*)d_in[9];   // [16,2048]
  const float* out_w    = (const float*)d_in[10];  // [1024,2048]
  float* out = (float*)d_out;                      // [2,2048,1024]

  // workspace layout (floats)
  float* xz = (float*)d_ws;                        // [4096,4096]: xc|z, later y|z
  float* u  = xz + (size_t)4096 * 4096;            // [4096,2048]
  float* dt = u  + (size_t)4096 * 2048;            // [4096,2048]
  float* bm = dt + (size_t)4096 * 2048;            // [4096,16]
  float* cm = bm + (size_t)4096 * 16;              // [4096,16]

  dim3 blk(256);
  // 1) xz = x @ in_proj_w^T                     (M=4096, N=4096, K=1024)
  gemm_kernel<0, false><<<dim3(32, 64), blk, 0, stream>>>(x, in_projw, xz, nullptr,
                                                          4096, 4096, 1024, 1024, 4096);
  // 2) u = causal conv(xc) + conv_b             (4 shifted GEMM passes, K=4*2048)
  conv_gemm_kernel<<<dim3(32, 32), blk, 0, stream>>>(xz, conv_w, conv_b, u);
  // 3) dt = clip(sigmoid(u @ dt_w^T + dt_b))    (M=4096, N=2048, K=2048)
  gemm_kernel<2, false><<<dim3(32, 32), blk, 0, stream>>>(u, dt_w, dt, dt_b,
                                                          4096, 2048, 2048, 2048, 2048);
  // 4) Bm = u @ B_w^T, Cm = u @ C_w^T           (N=16, guarded)
  gemm_kernel<0, true><<<dim3(32, 1), blk, 0, stream>>>(u, B_w, bm, nullptr,
                                                        4096, 16, 2048, 2048, 16);
  gemm_kernel<0, true><<<dim3(32, 1), blk, 0, stream>>>(u, C_w, cm, nullptr,
                                                        4096, 16, 2048, 2048, 16);
  // 5) selective scan + SiLU gate -> y into xz[:, 0:2048)
  scan_kernel<<<16, 256, 0, stream>>>(u, dt, bm, cm, A_log, Dv, xz);
  // 6) out = y @ out_w^T                        (M=4096, N=1024, K=2048, lda=4096)
  gemm_kernel<0, false><<<dim3(32, 16), blk, 0, stream>>>(xz, out_w, out, nullptr,
                                                          4096, 1024, 2048, 4096, 1024);
}
